// Proxy_42314017800609
// MI455X (gfx1250) — matrix operand order repaired and checked by reading, not compiled, not run on hardware
//
#include <hip/hip_runtime.h>
#include <math.h>

// ---- CDNA5 vector types ----
typedef __attribute__((ext_vector_type(16))) _Float16 v16h;
typedef __attribute__((ext_vector_type(8)))  float    v8f;
typedef __attribute__((ext_vector_type(4)))  float    f4;
typedef __attribute__((ext_vector_type(4)))  _Float16 h4;
typedef __attribute__((ext_vector_type(4)))  unsigned int u32x4;
typedef __attribute__((ext_vector_type(8)))  int i32x8;
typedef __attribute__((ext_vector_type(4)))  int i32x4;

#define D   512
#define NP  16384
#define NU  65536
#define NN  32768
#define PX  1024
#define KC  4
#define ALPHA_ 20.0f
#define TEMP_  10.0f

// ---- workspace layout (float element offsets). Requires ws_size >= ~340 MB. ----
#define IDX_POSN   0         // 512   normalized positive center
#define IDX_NEGC0  512       // 2048  normalized input negative centers
#define IDX_NEGCN  2560      // 2048  normalized updated negative centers
#define IDX_COUNTS 4608      // 4
#define IDX_NSUMS  4616      // 2048
#define IDX_SCAL   6664      // 32: [0]l_acc [1]rep [2]n_acc [3]Ssmall [4]pseudo_acc [5]Sbig [6]u_acc
#define IDX_RSS    6700      // 4     RS_small
#define IDX_RSM    6704      // 4     r_small
#define IDX_RBIG   7168      // 1024  r_big
#define IDX_RSBIG  8192      // 1024  RS_big
#define IDX_PROXH  9216      // half[1024*512]   (262144 floats)
#define IDX_SIMS   271360    // float[65536*4]
#define IDX_UNH    533504    // half[65536*512]  (16777216 floats)
#define IDX_SIMB   17310720  // float[65536*1024]

__device__ inline float wred_add(float v){
#pragma unroll
  for(int o=16;o;o>>=1) v += __shfl_xor(v,o,32);
  return v;
}
__device__ inline float wred_max(float v){
#pragma unroll
  for(int o=16;o;o>>=1) v = fmaxf(v,__shfl_xor(v,o,32));
  return v;
}

// ---------------- init: zero accumulators, normalize pos + input centers ----------------
__global__ void k_init(float* W, const float* pos, const float* negc){
  int t = threadIdx.x;
  for(int i = IDX_COUNTS + t; i < IDX_RSBIG + 1024; i += blockDim.x) W[i] = 0.f;
  __syncthreads();
  int wid = t>>5, lane = t&31;
  if(wid < 5){
    const float* src = (wid==0) ? pos : (negc + (size_t)(wid-1)*D);
    float* dst = (wid==0) ? (W+IDX_POSN) : (W+IDX_NEGC0+(wid-1)*D);
    float vals[16]; float ss = 0.f;
#pragma unroll
    for(int j=0;j<16;j++){ int c=j*32+lane; float v=src[c]; vals[j]=v; ss+=v*v; }
    ss = wred_add(ss);
    float inv = 1.f/fmaxf(sqrtf(ss), 1e-12f);
#pragma unroll
    for(int j=0;j<16;j++){ int c=j*32+lane; dst[c]=vals[j]*inv; }
  }
}

// ---------------- l_loss: mean(1 - pos_n . p_norm) ----------------
__global__ void k_lloss(float* W, const float* p){
  __shared__ float pos_s[D];
  __shared__ float bacc;
  int t = threadIdx.x;
  for(int i=t;i<D;i+=blockDim.x) pos_s[i]=W[IDX_POSN+i];
  if(t==0) bacc = 0.f;
  __syncthreads();
  int wid=t>>5, lane=t&31;
  float lacc = 0.f;
  for(int row = blockIdx.x*8 + wid; row < NP; row += gridDim.x*8){
    const float* pr = p + (size_t)row*D;
    float ss=0.f, dp=0.f;
#pragma unroll
    for(int j=0;j<4;j++){
      int c=j*128+lane*4;
      f4 v=*(const f4*)(pr+c);
      f4 q=*(const f4*)(pos_s+c);
#pragma unroll
      for(int k=0;k<4;k++){ ss+=v[k]*v[k]; dp+=v[k]*q[k]; }
    }
    ss=wred_add(ss); dp=wred_add(dp);
    if(lane==0){
      float inv=1.f/fmaxf(sqrtf(ss),1e-12f);
      lacc += 1.f - dp*inv;
    }
  }
  if((t&31)==0) atomicAdd(&bacc, lacc);
  __syncthreads();
  if(t==0) atomicAdd(&W[IDX_SCAL+0], bacc);
}

// ---------------- n_feats pass 1: cluster assignment + masked sums ----------------
__global__ void k_nassign(float* W, const float* nf){
  __shared__ float pos_s[D];
  __shared__ float c_s[KC*D];
  __shared__ float ns_s[KC*D];
  __shared__ float cnt_s[KC];
  int t=threadIdx.x;
  for(int i=t;i<D;i+=blockDim.x) pos_s[i]=W[IDX_POSN+i];
  for(int i=t;i<KC*D;i+=blockDim.x){ c_s[i]=W[IDX_NEGC0+i]; ns_s[i]=0.f; }
  if(t<KC) cnt_s[t]=0.f;
  __syncthreads();
  int wid=t>>5, lane=t&31;
  for(int row = blockIdx.x*8+wid; row<NN; row += gridDim.x*8){
    const float* pr = nf + (size_t)row*D;
    f4 v[4];
    float ss=0,dp=0,d0=0,d1=0,d2=0,d3=0;
#pragma unroll
    for(int j=0;j<4;j++){
      int c=j*128+lane*4;
      v[j]=*(const f4*)(pr+c);
      f4 q=*(const f4*)(pos_s+c);
      f4 a=*(const f4*)(c_s+0*D+c);
      f4 b=*(const f4*)(c_s+1*D+c);
      f4 e=*(const f4*)(c_s+2*D+c);
      f4 g=*(const f4*)(c_s+3*D+c);
#pragma unroll
      for(int k=0;k<4;k++){
        float x=v[j][k];
        ss+=x*x; dp+=x*q[k]; d0+=x*a[k]; d1+=x*b[k]; d2+=x*e[k]; d3+=x*g[k];
      }
    }
    ss=wred_add(ss); dp=wred_add(dp);
    d0=wred_add(d0); d1=wred_add(d1); d2=wred_add(d2); d3=wred_add(d3);
    float inv=1.f/fmaxf(sqrtf(ss),1e-12f);
    int keep = (dp*inv) < 0.7f;
    int am=0; float bv=d0;
    if(d1>bv){bv=d1;am=1;} if(d2>bv){bv=d2;am=2;} if(d3>bv){bv=d3;am=3;}
    if(keep){
      if(lane==0) atomicAdd(&cnt_s[am],1.f);
#pragma unroll
      for(int j=0;j<4;j++){
        int c=j*128+lane*4;
        atomicAdd(&ns_s[am*D+c+0], v[j][0]);
        atomicAdd(&ns_s[am*D+c+1], v[j][1]);
        atomicAdd(&ns_s[am*D+c+2], v[j][2]);
        atomicAdd(&ns_s[am*D+c+3], v[j][3]);
      }
    }
  }
  __syncthreads();
  if(t<KC) atomicAdd(&W[IDX_COUNTS+t], cnt_s[t]);
  for(int i=t;i<KC*D;i+=blockDim.x) atomicAdd(&W[IDX_NSUMS+i], ns_s[i]);
}

// ---------------- EMA center update + repulsion (single block) ----------------
__global__ void k_finalize(float* W, const float* negc_in){
  __shared__ float ncn[KC*D];
  __shared__ float snp[KC];
  __shared__ float snn[16];
  int t=threadIdx.x, wid=t>>5, lane=t&31;
  if(wid<KC){
    int k=wid;
    float cnt = W[IDX_COUNTS+k];
    float cm = fmaxf(cnt,1.f);
    int sel = cnt >= 3.f;
    float m[16], in[16]; float ss=0;
#pragma unroll
    for(int j=0;j<16;j++){
      int c=j*32+lane;
      m[j]=W[IDX_NSUMS+k*D+c]/cm;
      in[j]=negc_in[(size_t)k*D+c];
      ss+=m[j]*m[j];
    }
    ss=wred_add(ss);
    float inv1=1.f/fmaxf(sqrtf(ss),1e-12f);
    float ss2=0;
#pragma unroll
    for(int j=0;j<16;j++){ m[j]=in[j]+(m[j]*inv1-in[j])*0.1f; ss2+=m[j]*m[j]; }
    ss2=wred_add(ss2);
    float inv2=1.f/fmaxf(sqrtf(ss2),1e-12f);
    float ss3=0;
#pragma unroll
    for(int j=0;j<16;j++){ float f = sel ? m[j]*inv2 : in[j]; m[j]=f; ss3+=f*f; }
    ss3=wred_add(ss3);
    float inv3=1.f/fmaxf(sqrtf(ss3),1e-12f);
#pragma unroll
    for(int j=0;j<16;j++){
      int c=j*32+lane;
      float f=m[j]*inv3;
      ncn[k*D+c]=f; W[IDX_NEGCN+k*D+c]=f;
    }
  }
  __syncthreads();
  if(wid<KC){
    float d=0;
#pragma unroll
    for(int j=0;j<16;j++){ int c=j*32+lane; d+=ncn[wid*D+c]*W[IDX_POSN+c]; }
    d=wred_add(d);
    if(lane==0) snp[wid]=d;
  }
  for(int p=wid*2; p<wid*2+2; p++){
    int k1=p>>2, k2=p&3;
    float d=0;
#pragma unroll
    for(int j=0;j<16;j++){ int c=j*32+lane; d+=ncn[k1*D+c]*ncn[k2*D+c]; }
    d=wred_add(d);
    if(lane==0) snn[p]=d;
  }
  __syncthreads();
  if(t==0){
    float rnp=0;
    for(int k=0;k<KC;k++) rnp += logf(fmaxf(1.f-snp[k],1e-6f));
    rnp = -rnp/(float)KC;
    float rnn=0;
    for(int p=0;p<16;p++) if((p>>2)!=(p&3)) rnn += logf(fmaxf(1.f-snn[p],1e-6f));
    rnn = -rnn/(float)(KC*KC-KC);
    W[IDX_SCAL+1] = rnp + 0.1f*rnn;
  }
}

// ---------------- n branch: entropy of softmax(sim_n) ----------------
__global__ void k_nloss(float* W, const float* nf){
  __shared__ float c_s[KC*D];
  __shared__ float bacc;
  int t=threadIdx.x;
  for(int i=t;i<KC*D;i+=blockDim.x) c_s[i]=W[IDX_NEGCN+i];
  if(t==0) bacc=0.f;
  __syncthreads();
  int wid=t>>5, lane=t&31;
  float lacc=0.f;
  for(int row=blockIdx.x*8+wid; row<NN; row+=gridDim.x*8){
    const float* pr=nf+(size_t)row*D;
    float ss=0,d0=0,d1=0,d2=0,d3=0;
#pragma unroll
    for(int j=0;j<4;j++){
      int c=j*128+lane*4;
      f4 v=*(const f4*)(pr+c);
      f4 a=*(const f4*)(c_s+0*D+c), b=*(const f4*)(c_s+1*D+c),
         e=*(const f4*)(c_s+2*D+c), g=*(const f4*)(c_s+3*D+c);
#pragma unroll
      for(int k=0;k<4;k++){
        float x=v[k];
        ss+=x*x; d0+=x*a[k]; d1+=x*b[k]; d2+=x*e[k]; d3+=x*g[k];
      }
    }
    ss=wred_add(ss);
    d0=wred_add(d0); d1=wred_add(d1); d2=wred_add(d2); d3=wred_add(d3);
    if(lane==0){
      float inv=1.f/fmaxf(sqrtf(ss),1e-12f);
      float s0=d0*inv,s1=d1*inv,s2=d2*inv,s3=d3*inv;
      float m=fmaxf(fmaxf(s0,s1),fmaxf(s2,s3));
      float e0=__expf(s0-m),e1=__expf(s1-m),e2=__expf(s2-m),e3=__expf(s3-m);
      float z=e0+e1+e2+e3;
      float sp=e0*(s0-m)+e1*(s1-m)+e2*(s2-m)+e3*(s3-m);
      lacc += __logf(z) - sp/z;
    }
  }
  if((t&31)==0) atomicAdd(&bacc,lacc);
  __syncthreads();
  if(t==0) atomicAdd(&W[IDX_SCAL+2],bacc);
}

// ---------------- u vs neg centers: sim_small + S_small ----------------
__global__ void k_usmall(float* W, const float* uf){
  __shared__ float c_s[KC*D];
  __shared__ float bacc;
  int t=threadIdx.x;
  for(int i=t;i<KC*D;i+=blockDim.x) c_s[i]=W[IDX_NEGCN+i];
  if(t==0) bacc=0.f;
  __syncthreads();
  int wid=t>>5, lane=t&31;
  float lacc=0.f;
  for(int row=blockIdx.x*8+wid; row<NU; row+=gridDim.x*8){
    const float* pr=uf+(size_t)row*D;
    float ss=0,d0=0,d1=0,d2=0,d3=0;
#pragma unroll
    for(int j=0;j<4;j++){
      int c=j*128+lane*4;
      f4 v=*(const f4*)(pr+c);
      f4 a=*(const f4*)(c_s+0*D+c), b=*(const f4*)(c_s+1*D+c),
         e=*(const f4*)(c_s+2*D+c), g=*(const f4*)(c_s+3*D+c);
#pragma unroll
      for(int k=0;k<4;k++){
        float x=v[k];
        ss+=x*x; d0+=x*a[k]; d1+=x*b[k]; d2+=x*e[k]; d3+=x*g[k];
      }
    }
    ss=wred_add(ss);
    d0=wred_add(d0); d1=wred_add(d1); d2=wred_add(d2); d3=wred_add(d3);
    if(lane==0){
      float inv=1.f/fmaxf(sqrtf(ss),1e-12f);
      f4 s; s[0]=d0*inv; s[1]=d1*inv; s[2]=d2*inv; s[3]=d3*inv;
      *(f4*)(W+IDX_SIMS+(size_t)row*4)=s;
      lacc += __expf(ALPHA_*s[0])+__expf(ALPHA_*s[1])+__expf(ALPHA_*s[2])+__expf(ALPHA_*s[3]);
    }
  }
  if((t&31)==0) atomicAdd(&bacc,lacc);
  __syncthreads();
  if(t==0) atomicAdd(&W[IDX_SCAL+3],bacc);
}

// ---------------- small sinkhorn: fused c + rowsum pass ----------------
__global__ void k_srow_small(float* W, int first){
  __shared__ float red[KC];
  int t=threadIdx.x;
  if(t<KC) red[t]=0.f;
  __syncthreads();
  float S=W[IDX_SCAL+3];
  float r0=W[IDX_RSM+0],r1=W[IDX_RSM+1],r2=W[IDX_RSM+2],r3=W[IDX_RSM+3];
  float a0=0,a1=0,a2=0,a3=0;
  for(int b=blockIdx.x*blockDim.x+t; b<NU; b+=gridDim.x*blockDim.x){
    f4 s=*(const f4*)(W+IDX_SIMS+(size_t)b*4);
    float e0=__expf(ALPHA_*s[0]),e1=__expf(ALPHA_*s[1]),e2=__expf(ALPHA_*s[2]),e3=__expf(ALPHA_*s[3]);
    float cb=1.f;
    if(!first){
      float rd=e0*r0+e1*r1+e2*r2+e3*r3;
      cb = S/(rd*(float)NU);
    }
    a0+=e0*cb; a1+=e1*cb; a2+=e2*cb; a3+=e3*cb;
  }
  atomicAdd(&red[0],a0); atomicAdd(&red[1],a1);
  atomicAdd(&red[2],a2); atomicAdd(&red[3],a3);
  __syncthreads();
  if(t<KC) atomicAdd(&W[IDX_RSS+t],red[t]);
}

__global__ void k_rupd_small(float* W){
  int t=threadIdx.x;
  if(t<KC){
    float S=W[IDX_SCAL+3];
    float rs=W[IDX_RSS+t];
    W[IDX_RSM+t]=S/((float)KC*rs);
    W[IDX_RSS+t]=0.f;
  }
}

// ---------------- loss_pseudo ----------------
__global__ void k_uloss_small(float* W){
  __shared__ float bacc;
  int t=threadIdx.x;
  if(t==0) bacc=0.f;
  __syncthreads();
  float r0=W[IDX_RSM+0],r1=W[IDX_RSM+1],r2=W[IDX_RSM+2],r3=W[IDX_RSM+3];
  float lacc=0.f;
  for(int b=blockIdx.x*blockDim.x+t; b<NU; b+=gridDim.x*blockDim.x){
    f4 s=*(const f4*)(W+IDX_SIMS+(size_t)b*4);
    float e0=__expf(ALPHA_*s[0]),e1=__expf(ALPHA_*s[1]),e2=__expf(ALPHA_*s[2]),e3=__expf(ALPHA_*s[3]);
    float rd=e0*r0+e1*r1+e2*r2+e3*r3;
    float inv=1.f/rd;
    float term=(e0*r0*s[0]+e1*r1*s[1]+e2*r2*s[2]+e3*r3*s[3])*inv;
    float m=fmaxf(fmaxf(s[0],s[1]),fmaxf(s[2],s[3]));
    float z=__expf(s[0]-m)+__expf(s[1]-m)+__expf(s[2]-m)+__expf(s[3]-m);
    lacc += (m+__logf(z)) - term;
  }
  atomicAdd(&bacc,lacc);
  __syncthreads();
  if(t==0) atomicAdd(&W[IDX_SCAL+4],bacc);
}

// ---------------- normalize + convert to f16 ----------------
__global__ void k_proxyprep(float* W, const float* px){
  _Float16* Ph=(_Float16*)(W+IDX_PROXH);
  int t=threadIdx.x, wid=t>>5, lane=t&31;
  for(int row=blockIdx.x*8+wid; row<PX; row+=gridDim.x*8){
    const float* pr=px+(size_t)row*D;
    f4 v[4]; float ss=0;
#pragma unroll
    for(int j=0;j<4;j++){
      int c=j*128+lane*4;
      v[j]=*(const f4*)(pr+c);
#pragma unroll
      for(int k=0;k<4;k++) ss+=v[j][k]*v[j][k];
    }
    ss=wred_add(ss);
    float inv=1.f/fmaxf(sqrtf(ss),1e-12f);
#pragma unroll
    for(int j=0;j<4;j++){
      int c=j*128+lane*4;
      h4 h;
#pragma unroll
      for(int k=0;k<4;k++) h[k]=(_Float16)(v[j][k]*inv);
      *(h4*)(Ph+(size_t)row*D+c)=h;
    }
  }
}

__global__ void k_uprep(float* W, const float* uf){
  _Float16* Uh=(_Float16*)(W+IDX_UNH);
  int t=threadIdx.x, wid=t>>5, lane=t&31;
  for(int row=blockIdx.x*8+wid; row<NU; row+=gridDim.x*8){
    const float* pr=uf+(size_t)row*D;
    f4 v[4]; float ss=0;
#pragma unroll
    for(int j=0;j<4;j++){
      int c=j*128+lane*4;
      v[j]=*(const f4*)(pr+c);
#pragma unroll
      for(int k=0;k<4;k++) ss+=v[j][k]*v[j][k];
    }
    ss=wred_add(ss);
    float inv=1.f/fmaxf(sqrtf(ss),1e-12f);
#pragma unroll
    for(int j=0;j<4;j++){
      int c=j*128+lane*4;
      h4 h;
#pragma unroll
      for(int k=0;k<4;k++) h[k]=(_Float16)(v[j][k]*inv);
      *(h4*)(Uh+(size_t)row*D+c)=h;
    }
  }
}

// ---------------- WMMA GEMM: sim_big[65536x1024] = Un @ Pn^T ----------------
// Block tile 256(M) x 64(N). B strip (64 proxies x 512 dims, f16 = 64KB) is
// DMA'd into LDS once per block by the Tensor Data Mover, then the K-loop
// reads B via ds_load_b128 and streams A from global (prefetched).
//  A lane<16: row=lane, K {d0+0..7, d0+16..23}; lane>=16: row=lane-16, K {d0+8..15, d0+24..31}
//  B lane<16: col=lane, K {d0+0..7, d0+16..23}; lane>=16: col=lane-16, K {d0+8..15, d0+24..31}
__global__ void k_gemm(float* W){
  const _Float16* Uh=(const _Float16*)(W+IDX_UNH);
  const _Float16* Ph=(const _Float16*)(W+IDX_PROXH);
  float* SB=W+IDX_SIMB;
  __shared__ _Float16 Bsh[64*D];   // 64KB
  int t=threadIdx.x, wid=t>>5, lane=t&31;
  int m0=blockIdx.y*256 + wid*16;
  int n0=blockIdx.x*64;

  if(wid==0){
    // ---- Tensor DMA descriptor: 2D tile, 64 rows x 512 halves, row stride 512 ----
    unsigned long long gaddr = (unsigned long long)(const void*)(Ph + (size_t)n0*D);
    unsigned lds  = (unsigned)(unsigned long long)(const void*)&Bsh[0];
    u32x4 g0;
    g0[0] = 1u;                                   // count=1 valid, user mode
    g0[1] = lds;                                  // lds_addr
    g0[2] = (unsigned)(gaddr & 0xFFFFFFFFull);    // global_addr[31:0]
    g0[3] = (unsigned)((gaddr>>32) & 0x01FFFFFFu) | (2u<<30); // addr[56:32] | type=2
    i32x8 g1;
    g1[0] = (1<<16);                              // data_size=1 (2 bytes)
    g1[1] = (int)((unsigned)(D & 0xFFFF) << 16);  // tensor_dim0 low16 -> [63:48]
    g1[2] = (int)((D>>16) | (64u<<16));           // tensor_dim0 hi | tensor_dim1 low16
    g1[3] = (int)((64u>>16) | ((unsigned)D<<16)); // tensor_dim1 hi | tile_dim0=512
    g1[4] = 64;                                   // tile_dim1=64, tile_dim2=0
    g1[5] = D;                                    // tensor_dim0_stride=512 (low32)
    g1[6] = 0;
    g1[7] = 0;
    i32x4 gz4 = {0,0,0,0};
    i32x8 gz8 = {0,0,0,0,0,0,0,0};
    __builtin_amdgcn_tensor_load_to_lds(g0, g1, gz4, gz4, gz8, 0);
    __builtin_amdgcn_s_wait_tensorcnt(0);
  }
  __syncthreads();

  int hiA=(lane&16)?8:0;
  int hiB=(lane&16)?8:0;
  const _Float16* aptr0 = Uh + (size_t)(m0+(lane&15))*D + hiA;
  const _Float16* aptr1 = aptr0 + (size_t)128*D;
  const _Float16* bsh0  = Bsh + (size_t)(lane&15)*D + hiB;

  v8f acc[8];
#pragma unroll
  for(int i=0;i<8;i++){ v8f z={}; acc[i]=z; }

  for(int d0=0; d0<D; d0+=32){
    __builtin_prefetch((const void*)(aptr0+d0+32), 0, 0);
    __builtin_prefetch((const void*)(aptr1+d0+32), 0, 0);
    union { v16h h; f4 f[2]; } A0, A1;
    A0.f[0]=*(const f4*)(aptr0+d0);
    A0.f[1]=*(const f4*)(aptr0+d0+16);
    A1.f[0]=*(const f4*)(aptr1+d0);
    A1.f[1]=*(const f4*)(aptr1+d0+16);
#pragma unroll
    for(int tt=0;tt<4;tt++){
      union { v16h h; f4 f[2]; } B;
      const _Float16* bp = bsh0 + (size_t)tt*16*D + d0;
      B.f[0]=*(const f4*)(bp);
      B.f[1]=*(const f4*)(bp+16);
      acc[tt]  =__builtin_amdgcn_wmma_f32_16x16x32_f16(false, A0.h, false, B.h,
                                                       (short)0, acc[tt],   false, false);
      acc[4+tt]=__builtin_amdgcn_wmma_f32_16x16x32_f16(false, A1.h, false, B.h,
                                                       (short)0, acc[4+tt], false, false);
    }
  }
  int cnb=lane&15;
  int cm0=m0+((lane&16)?8:0);
#pragma unroll
  for(int tt=0;tt<4;tt++){
    int cn=n0+tt*16+cnb;
#pragma unroll
    for(int r=0;r<8;r++){
      SB[(size_t)(cm0+r)*PX+cn]     = acc[tt][r];
      SB[(size_t)(cm0+128+r)*PX+cn] = acc[4+tt][r];
    }
  }
}

// ---------------- big sinkhorn: fused c + rowsum pass (wave per row) ----------------
__global__ void k_srow_big(float* W, int first){
  __shared__ float r_s[PX];
  __shared__ float colacc[8][PX];
  int t=threadIdx.x, wid=t>>5, lane=t&31;
  for(int i=t;i<PX;i+=256) r_s[i]=W[IDX_RBIG+i];
  for(int i=t;i<8*PX;i+=256) ((float*)colacc)[i]=0.f;
  __syncthreads();
  const float* SB=W+IDX_SIMB;
  float S=W[IDX_SCAL+5];
  float* ca=colacc[wid];
  for(int b=blockIdx.x*8+wid; b<NU; b+=gridDim.x*8){
    const float* rp=SB+(size_t)b*PX;
    f4 ev[8];
    float rd=0.f;
#pragma unroll
    for(int j=0;j<8;j++){
      int c=j*128+lane*4;
      f4 s=*(const f4*)(rp+c);
      f4 e;
      e[0]=__expf(ALPHA_*s[0]); e[1]=__expf(ALPHA_*s[1]);
      e[2]=__expf(ALPHA_*s[2]); e[3]=__expf(ALPHA_*s[3]);
      ev[j]=e;
      if(!first) rd += e[0]*r_s[c]+e[1]*r_s[c+1]+e[2]*r_s[c+2]+e[3]*r_s[c+3];
    }
    float cb=1.f;
    if(!first){ rd=wred_add(rd); cb=S/(rd*(float)NU); }
#pragma unroll
    for(int j=0;j<8;j++){
      int c=j*128+lane*4;
      ca[c]+=ev[j][0]*cb; ca[c+1]+=ev[j][1]*cb;
      ca[c+2]+=ev[j][2]*cb; ca[c+3]+=ev[j][3]*cb;
    }
  }
  __syncthreads();
  for(int i=t;i<PX;i+=256){
    float tot=0.f;
#pragma unroll
    for(int w=0;w<8;w++) tot+=colacc[w][i];
    atomicAdd(&W[IDX_RSBIG+i],tot);
  }
}

__global__ void k_rupd_big(float* W, int first){
  __shared__ float red[32];
  __shared__ float sS;
  int t=threadIdx.x, wid=t>>5, lane=t&31;
  float rs=W[IDX_RSBIG+t];
  if(first){
    float v=wred_add(rs);
    if(lane==0) red[wid]=v;
    __syncthreads();
    if(wid==0){
      float x=red[lane];
      x=wred_add(x);
      if(lane==0){ sS=x; W[IDX_SCAL+5]=x; }
    }
  } else {
    if(t==0) sS=W[IDX_SCAL+5];
  }
  __syncthreads();
  float S=sS;
  W[IDX_RBIG+t]=S/((float)PX*rs);
  W[IDX_RSBIG+t]=0.f;
}

// ---------------- u_loss: t = E*r/(E.r); lse over 10*sim ----------------
__global__ void k_uloss_big(float* W){
  __shared__ float r_s[PX];
  __shared__ float bacc;
  int t=threadIdx.x, wid=t>>5, lane=t&31;
  for(int i=t;i<PX;i+=256) r_s[i]=W[IDX_RBIG+i];
  if(t==0) bacc=0.f;
  __syncthreads();
  const float* SB=W+IDX_SIMB;
  float lacc=0.f;
  for(int b=blockIdx.x*8+wid; b<NU; b+=gridDim.x*8){
    const float* rp=SB+(size_t)b*PX;
    f4 sv[8];
    float rd=0.f, mx=-1e30f;
#pragma unroll
    for(int j=0;j<8;j++){
      int c=j*128+lane*4;
      f4 s=*(const f4*)(rp+c);
      sv[j]=s;
#pragma unroll
      for(int k=0;k<4;k++){
        float e=__expf(ALPHA_*s[k]);
        rd += e*r_s[c+k];
        mx = fmaxf(mx,s[k]);
      }
    }
    rd=wred_add(rd); mx=wred_max(mx);
    float inv=1.f/rd;
    float term=0.f, z=0.f;
#pragma unroll
    for(int j=0;j<8;j++){
      int c=j*128+lane*4;
      f4 s=sv[j];
#pragma unroll
      for(int k=0;k<4;k++){
        float e=__expf(ALPHA_*s[k]);
        term += (e*r_s[c+k]*inv)*(TEMP_*s[k]);
        z += __expf(TEMP_*(s[k]-mx));
      }
    }
    term=wred_add(term); z=wred_add(z);
    if(lane==0) lacc += (TEMP_*mx + __logf(z)) - term;
  }
  if((t&31)==0) atomicAdd(&bacc,lacc);
  __syncthreads();
  if(t==0) atomicAdd(&W[IDX_SCAL+6],bacc);
}

// ---------------- combine ----------------
__global__ void k_final(const float* W, float* out){
  float l   = W[IDX_SCAL+0]/(float)NP;
  float rep = W[IDX_SCAL+1];
  float ln  = W[IDX_SCAL+2]/(float)NN;
  float lp  = W[IDX_SCAL+4]/(float)NU;
  float nl  = 0.5f*(ln+lp);
  float ul  = W[IDX_SCAL+6]/(float)NU;
  float cls = 0.45f*l + 0.3f*nl + 0.10f*ul + 0.15f*rep;
  out[0]=cls; out[1]=l; out[2]=ul; out[3]=nl; out[4]=rep;
}

extern "C" void kernel_launch(void* const* d_in, const int* in_sizes, int n_in,
                              void* d_out, int out_size, void* d_ws, size_t ws_size,
                              hipStream_t stream){
  (void)in_sizes; (void)n_in; (void)out_size; (void)ws_size;
  const float* p   = (const float*)d_in[0];
  const float* u   = (const float*)d_in[1];
  const float* nf  = (const float*)d_in[2];
  const float* pos = (const float*)d_in[3];
  const float* px  = (const float*)d_in[4];
  const float* nc  = (const float*)d_in[5];
  float* W = (float*)d_ws;

  k_init<<<1,1024,0,stream>>>(W,pos,nc);
  k_lloss<<<128,256,0,stream>>>(W,p);
  k_nassign<<<128,256,0,stream>>>(W,nf);
  k_finalize<<<1,256,0,stream>>>(W,nc);
  k_nloss<<<128,256,0,stream>>>(W,nf);
  k_usmall<<<256,256,0,stream>>>(W,u);
  k_srow_small<<<256,256,0,stream>>>(W,1);
  k_rupd_small<<<1,32,0,stream>>>(W);
  k_srow_small<<<256,256,0,stream>>>(W,0);
  k_rupd_small<<<1,32,0,stream>>>(W);
  k_srow_small<<<256,256,0,stream>>>(W,0);
  k_rupd_small<<<1,32,0,stream>>>(W);
  k_uloss_small<<<256,256,0,stream>>>(W);
  k_proxyprep<<<128,256,0,stream>>>(W,px);
  k_uprep<<<1024,256,0,stream>>>(W,u);
  k_gemm<<<dim3(16,256),256,0,stream>>>(W);
  k_srow_big<<<512,256,0,stream>>>(W,1);
  k_rupd_big<<<1,1024,0,stream>>>(W,1);
  k_srow_big<<<512,256,0,stream>>>(W,0);
  k_rupd_big<<<1,1024,0,stream>>>(W,0);
  k_srow_big<<<512,256,0,stream>>>(W,0);
  k_rupd_big<<<1,1024,0,stream>>>(W,0);
  k_uloss_big<<<512,256,0,stream>>>(W);
  k_final<<<1,1,0,stream>>>(W,(float*)d_out);
}
